// ResidualGN_35820027248939
// MI455X (gfx1250) — matrix operand order repaired and checked by reading, not compiled, not run on hardware
//
#include <hip/hip_runtime.h>
#include <hip/hip_bf16.h>
#include <cstddef>

// ---------------------------------------------------------------------------
// Problem constants (match the reference)
// ---------------------------------------------------------------------------
#define NRES   65536      // N = B*L
#define LCH    1024       // residues per chain
#define BCH    64         // chains
#define HDIM   128
#define CDIM   256
#define HCDIM  64

typedef __attribute__((ext_vector_type(16))) _Float16 v16h;
typedef __attribute__((ext_vector_type(8)))  _Float16 v8h;
typedef __attribute__((ext_vector_type(4)))  _Float16 v4h;
typedef __attribute__((ext_vector_type(8)))  float    v8f;
typedef __attribute__((ext_vector_type(4)))  float    v4f;

// ---------------------------------------------------------------------------
// One-time per-call weight conversion: f32 W[m*so + k*si + tap*st]
//   -> f16 Wt[tap][m][k]  (so B fragments are two runs of 8 consecutive f16 = b128 loads)
// ---------------------------------------------------------------------------
__global__ void weight_prep_kernel(const float* __restrict__ W,
                                   _Float16* __restrict__ Wt,
                                   int M, int K, int so, int si, int st, int ntaps)
{
  const int i = blockIdx.x * blockDim.x + threadIdx.x;
  const int tot = ntaps * M * K;
  if (i >= tot) return;
  const int tap = i / (M * K);
  const int rem = i - tap * M * K;
  const int m = rem / K;
  const int k = rem - m * K;
  Wt[i] = (_Float16)W[(size_t)m * so + (size_t)k * si + (size_t)tap * st];
}

// ---------------------------------------------------------------------------
// Templated WMMA GEMM over node-major activations:
//   Y(N x M) = [gn(relu(X))] (N x K) @ Wt(K x M, f16) (+ taps, + bias, +resid)
// NTAPS==3 implements the dilated k=3 conv with tap offsets (-d, 0, +d) along
// L, zero-masked outside the chain (padding applied AFTER groupnorm, matching
// conv_general_dilated). GN-apply is fused into LDS staging (GN==true).
// One block = 16 output rows; one wave per 16-column tile.
//
// Fragment layouts (CDNA5 ISA §7.12.2, wave32):
//   A 16x32 f16 : lane row = lane&15 ; elem e -> K = (e<8 ? h*8+e : 16+h*8+e-8)
//   B 32x16 f16 : lane col = lane&15 ; elem e -> K = (e<8 ? h*8+e : 16+h*8+e-8)
//   C/D 16x16 f32: row = h*8 + r, col = lane&15           (h = lane>>4)
// ---------------------------------------------------------------------------
template <int K, int M, int NTAPS, bool GN, bool RESID>
__global__ __launch_bounds__((M / 16) * 32)
void gemm_wmma_kernel(const float* __restrict__ X,
                      const _Float16* __restrict__ Wt,   // [NTAPS][M][K]
                      const float* __restrict__ bias,    // [M]
                      const float* __restrict__ resid,   // [N][M] if RESID
                      float* __restrict__ Y,             // [N][M]
                      const float* __restrict__ mu,      // [BCH*32] if GN
                      const float* __restrict__ rs,
                      const float* __restrict__ gamma,   // [K]
                      const float* __restrict__ beta,
                      int dil)
{
  constexpr int THREADS = (M / 16) * 32;
  constexpr int CS = K / 32;                 // channels per GN group
  __shared__ _Float16 smemA[NTAPS * 16 * K];

  const int tid  = threadIdx.x;
  const int lane = tid & 31;
  const int wave = tid >> 5;
  const int row0 = blockIdx.x << 4;          // 16-row stripe
  const int col0 = wave << 4;
  const int bidx = row0 >> 10;               // chain index (L = 1024)

  // ---- cooperative A staging: float4 loads -> (relu,GN) -> 4xf16 stores ----
  constexpr int KQ = K / 4;
  for (int i = tid; i < NTAPS * 16 * KQ; i += THREADS) {
    const int tap = i / (16 * KQ);
    const int rem = i - tap * 16 * KQ;
    const int r   = rem / KQ;
    const int k4  = (rem - r * KQ) * 4;
    const int off = (tap - (NTAPS >> 1)) * dil;
    const int l   = ((row0 + r) & (LCH - 1)) + off;
    v4f v = {0.0f, 0.0f, 0.0f, 0.0f};
    if (l >= 0 && l < LCH) {
      v = *(const v4f*)&X[(size_t)(row0 + r + off) * K + k4];
      if (GN) {
#pragma unroll
        for (int c = 0; c < 4; ++c) {
          float x = fmaxf(v[c], 0.0f);
          const int g = (k4 + c) / CS;       // CS is a power of two -> shift
          x = (x - mu[bidx * 32 + g]) * rs[bidx * 32 + g];
          v[c] = x * gamma[k4 + c] + beta[k4 + c];
        }
      }
    }
    v4h h;
#pragma unroll
    for (int c = 0; c < 4; ++c) h[c] = (_Float16)v[c];
    *(v4h*)&smemA[(tap * 16 + r) * K + k4] = h;
  }
  __syncthreads();

  const int m16  = lane & 15;
  const int half = lane >> 4;
  const int col  = col0 + m16;
  v8f acc = {};

#pragma unroll
  for (int tap = 0; tap < NTAPS; ++tap) {
    const _Float16* arow = &smemA[(tap * 16 + m16) * K];
    const _Float16* brow = &Wt[((size_t)tap * M + col) * K];
#pragma unroll
    for (int kb = 0; kb < K; kb += 32) {
      const v8h alo = *(const v8h*)(arow + kb + half * 8);        // K: kb+h*8 ..+7
      const v8h ahi = *(const v8h*)(arow + kb + 16 + half * 8);   // K: kb+16+h*8 ..
      const v16h a = __builtin_shufflevector(alo, ahi,
          0, 1, 2, 3, 4, 5, 6, 7, 8, 9, 10, 11, 12, 13, 14, 15);
      const v8h blo = *(const v8h*)(brow + kb + half * 8);        // K: kb+h*8 ..+7
      const v8h bhi = *(const v8h*)(brow + kb + 16 + half * 8);   // K: kb+16+h*8 ..
      const v16h b = __builtin_shufflevector(blo, bhi,
          0, 1, 2, 3, 4, 5, 6, 7, 8, 9, 10, 11, 12, 13, 14, 15);
      acc = __builtin_amdgcn_wmma_f32_16x16x32_f16(
          /*neg_a=*/false, a, /*neg_b=*/false, b,
          /*c_mod=*/(short)0, acc, /*reuse_a=*/false, /*reuse_b=*/false);
    }
  }

  // ---- epilogue ----
  const float bv = bias[col];
#pragma unroll
  for (int r = 0; r < 8; ++r) {
    const int row = row0 + half * 8 + r;
    const size_t o = (size_t)row * M + col;
    float y = acc[r] + bv;
    if (RESID) y += resid[o];
    Y[o] = y;
  }
}

// ---------------------------------------------------------------------------
// GroupNorm statistics of relu(X) over (CS channels x L) per (chain, group).
// grid = BCH*32 blocks, block = 256 threads. Vector loads per row.
// ---------------------------------------------------------------------------
template <int CS>
__global__ void gn_stats_kernel(const float* __restrict__ X,
                                float* __restrict__ mu, float* __restrict__ rs)
{
  constexpr int C = CS * 32;
  __shared__ float s1[256], s2[256];
  const int b = blockIdx.x >> 5;
  const int g = blockIdx.x & 31;
  float a = 0.0f, q = 0.0f;
  for (int l = threadIdx.x; l < LCH; l += blockDim.x) {
    const float* p = &X[(size_t)(b * LCH + l) * C + g * CS];
    if (CS % 4 == 0) {
#pragma unroll
      for (int c = 0; c < CS; c += 4) {
        const v4f v = *(const v4f*)(p + c);
#pragma unroll
        for (int j = 0; j < 4; ++j) {
          const float x = fmaxf(v[j], 0.0f);
          a += x; q += x * x;
        }
      }
    } else {
#pragma unroll
      for (int c = 0; c < CS; ++c) {
        const float x = fmaxf(p[c], 0.0f);
        a += x; q += x * x;
      }
    }
  }
  s1[threadIdx.x] = a; s2[threadIdx.x] = q;
  __syncthreads();
  for (int s = 128; s > 0; s >>= 1) {
    if (threadIdx.x < s) {
      s1[threadIdx.x] += s1[threadIdx.x + s];
      s2[threadIdx.x] += s2[threadIdx.x + s];
    }
    __syncthreads();
  }
  if (threadIdx.x == 0) {
    constexpr float tot = (float)(CS * LCH);
    const float m = s1[0] / tot;
    const float v = s2[0] / tot - m * m;
    mu[blockIdx.x] = m;
    rs[blockIdx.x] = rsqrtf(v + 1e-5f);
  }
}

// ---------------------------------------------------------------------------
// feat = [ state + (sin|cos(angles) @ unl_w + unl_b) , state ]   (N x 256)
// ---------------------------------------------------------------------------
__global__ void feat_build_kernel(const float* __restrict__ angles,
                                  const float* __restrict__ state,
                                  const float* __restrict__ unl_w,
                                  const float* __restrict__ unl_b,
                                  float* __restrict__ feat)
{
  __shared__ float trig[6];
  const int n = blockIdx.x;
  const int h = threadIdx.x;
  if (h < 3)       trig[h] = sinf(angles[n * 3 + h]);
  else if (h < 6)  trig[h] = cosf(angles[n * 3 + (h - 3)]);
  __syncthreads();
  float u = unl_b[h];
#pragma unroll
  for (int j = 0; j < 6; ++j) u += trig[j] * unl_w[j * HDIM + h];
  const float s = state[(size_t)n * HDIM + h];
  feat[(size_t)n * CDIM + h]        = s + u;
  feat[(size_t)n * CDIM + HDIM + h] = s;
}

// ---------------------------------------------------------------------------
// angles += out @ look_w + look_b ; wrap via atan2(sin,cos);
// angles_clean = wrapped ; angles = wrapped + 0.1*noise_k.
// ---------------------------------------------------------------------------
__global__ void angles_update_kernel(const float* __restrict__ outb,
                                     const float* __restrict__ look_w,
                                     const float* __restrict__ look_b,
                                     const float* __restrict__ noise_k,
                                     float* __restrict__ angles,
                                     float* __restrict__ angles_clean)
{
  __shared__ float sh[3][128];
  const int n = blockIdx.x, t = threadIdx.x;
  const float o = outb[(size_t)n * HDIM + t];
#pragma unroll
  for (int j = 0; j < 3; ++j) sh[j][t] = o * look_w[t * 3 + j];
  __syncthreads();
  for (int s = 64; s > 0; s >>= 1) {
    if (t < s) {
#pragma unroll
      for (int j = 0; j < 3; ++j) sh[j][t] += sh[j][t + s];
    }
    __syncthreads();
  }
  if (t < 3) {
    float a = angles[n * 3 + t] + sh[t][0] + look_b[t];
    a = atan2f(sinf(a), cosf(a));
    angles_clean[n * 3 + t] = a;
    angles[n * 3 + t] = a + 0.1f * noise_k[n * 3 + t];
  }
}

// ---------------------------------------------------------------------------
// Sequential NeRF chain extension: 1 thread per chain (64 total).
// ---------------------------------------------------------------------------
__device__ __forceinline__ void unit3(float& x, float& y, float& z)
{
  const float inv = 1.0f / (sqrtf(x * x + y * y + z * z) + 1e-8f);
  x *= inv; y *= inv; z *= inv;
}

__global__ void nerf_chain_kernel(const float* __restrict__ ang,
                                  float* __restrict__ pos)
{
  const int b = blockIdx.x * blockDim.x + threadIdx.x;
  if (b >= BCH) return;
  const float BL[3] = {1.459f, 1.525f, 1.329f};
  const float BA[3] = {2.124f, 1.941f, 2.028f};
  float ax = 0.0f,   ay = 0.0f, az = 0.0f;
  float bx = 1.459f, by = 0.0f, bz = 0.0f;
  float cx = 2.0f,   cy = 1.4f, cz = 0.0f;
  for (int l = 0; l < LCH; ++l) {
    const size_t n = (size_t)b * LCH + l;
#pragma unroll
    for (int j = 0; j < 3; ++j) {
      const float chi = ang[n * 3 + j];
      float bcx = cx - bx, bcy = cy - by, bcz = cz - bz;
      unit3(bcx, bcy, bcz);
      const float ux = bx - ax, uy = by - ay, uz = bz - az;
      float nx = uy * bcz - uz * bcy;
      float ny = uz * bcx - ux * bcz;
      float nz = ux * bcy - uy * bcx;
      unit3(nx, ny, nz);
      const float mx = ny * bcz - nz * bcy;
      const float my = nz * bcx - nx * bcz;
      const float mz = nx * bcy - ny * bcx;
      const float ct = cosf(BA[j]), st = sinf(BA[j]);
      const float cc = cosf(chi),   sc = sinf(chi);
      const float px = cx + BL[j] * (-ct * bcx + st * (cc * mx + sc * nx));
      const float py = cy + BL[j] * (-ct * bcy + st * (cc * my + sc * ny));
      const float pz = cz + BL[j] * (-ct * bcz + st * (cc * mz + sc * nz));
      pos[(n * 3 + j) * 3 + 0] = px;
      pos[(n * 3 + j) * 3 + 1] = py;
      pos[(n * 3 + j) * 3 + 2] = pz;
      ax = bx; ay = by; az = bz;
      bx = cx; by = cy; bz = cz;
      cx = px; cy = py; cz = pz;
    }
  }
}

// ---------------------------------------------------------------------------
// Host orchestration.
// Input flattening assumption (jax sorted-key dict tree order):
//  0 inputs, 1 indices, 2 init_angles, 3 noise, 4 ang0_b, 5 ang0_w,
//  6.. blocks[0..2]: { ab, aw, hb, hw,
//        inner[0..3]: { b0,b1,b2, gn_b0..2, gn_g0..2, w0,w1,w2 } }  (52 each)
//  162 look_b, 163 look_w, 164 state_b, 165 state_w, 166 unl_b, 167 unl_w
// ---------------------------------------------------------------------------
extern "C" void kernel_launch(void* const* d_in, const int* in_sizes, int n_in,
                              void* d_out, int out_size, void* d_ws, size_t ws_size,
                              hipStream_t stream)
{
  const float* inputs      = (const float*)d_in[0];
  const float* init_angles = (const float*)d_in[2];
  const float* noise       = (const float*)d_in[3];

  struct Inner { const float *b[3], *gn_b[3], *gn_g[3], *w[3]; };
  struct Block { const float *ab, *aw, *hb, *hw; Inner inner[4]; };
  Block blk[3];
  int p = 6;
  for (int d = 0; d < 3; ++d) {
    blk[d].ab = (const float*)d_in[p++];
    blk[d].aw = (const float*)d_in[p++];
    blk[d].hb = (const float*)d_in[p++];
    blk[d].hw = (const float*)d_in[p++];
    for (int j = 0; j < 4; ++j) {
      Inner& in = blk[d].inner[j];
      for (int t = 0; t < 3; ++t) in.b[t]    = (const float*)d_in[p++];
      for (int t = 0; t < 3; ++t) in.gn_b[t] = (const float*)d_in[p++];
      for (int t = 0; t < 3; ++t) in.gn_g[t] = (const float*)d_in[p++];
      for (int t = 0; t < 3; ++t) in.w[t]    = (const float*)d_in[p++];
    }
  }
  const float* look_b  = (const float*)d_in[p++];
  const float* look_w  = (const float*)d_in[p++];
  const float* state_b = (const float*)d_in[p++];
  const float* state_w = (const float*)d_in[p++];
  const float* unl_b   = (const float*)d_in[p++];
  const float* unl_w   = (const float*)d_in[p++];

  // ---- workspace layout: f32 activations then f16 converted weights ----
  float* ws     = (float*)d_ws;
  float* state  = ws;                                    // N x 128
  float* feat   = state  + (size_t)NRES * HDIM;          // N x 256
  float* h1     = feat   + (size_t)NRES * CDIM;          // N x 64
  float* h2     = h1     + (size_t)NRES * HCDIM;         // N x 64
  float* outb   = h2     + (size_t)NRES * HCDIM;         // N x 128
  float* angles = outb   + (size_t)NRES * HDIM;          // N x 3
  float* aclean = angles + (size_t)NRES * 3;             // N x 3
  float* mu     = aclean + (size_t)NRES * 3;             // 2048
  float* rs     = mu + 2048;                             // 2048
  _Float16* wtb = (_Float16*)(rs + 2048);
  size_t wofs = 0;
  auto prep = [&](const float* W, int M, int K, int so, int si, int st,
                  int ntaps) -> const _Float16* {
    _Float16* dst = wtb + wofs;
    const int tot = ntaps * M * K;
    weight_prep_kernel<<<(tot + 255) / 256, 256, 0, stream>>>(
        W, dst, M, K, so, si, st, ntaps);
    wofs += (size_t)tot;
    return dst;
  };

  // convert all weights to f16 [tap][m][k] once per call
  const _Float16* wt_state = prep(state_w, HDIM, 64, 1, HDIM, 0, 1);
  const _Float16 *wt_aw[3], *wt_hw[3], *wt1[3][4], *wt2[3][4], *wt3[3][4];
  for (int d = 0; d < 3; ++d) {
    wt_aw[d] = prep(blk[d].aw, HDIM, CDIM, 1, HDIM, 0, 1);
    wt_hw[d] = prep(blk[d].hw, HDIM, CDIM, 1, HDIM, 0, 1);
    for (int j = 0; j < 4; ++j) {
      wt1[d][j] = prep(blk[d].inner[j].w[0], HCDIM, CDIM, CDIM, 1, 0, 1); // 1x1 256->64
      wt2[d][j] = prep(blk[d].inner[j].w[1], HCDIM, HCDIM, 192, 3, 1, 3); // k=3  64->64
      wt3[d][j] = prep(blk[d].inner[j].w[2], CDIM, HCDIM, HCDIM, 1, 0, 1);// 1x1  64->256
    }
  }

  const dim3 grows(NRES / 16);

  // angles <- init_angles (must not mutate input buffers)
  hipMemcpyAsync(angles, init_angles, (size_t)NRES * 3 * sizeof(float),
                 hipMemcpyDeviceToDevice, stream);

  // state = inputs @ state_w + state_b
  gemm_wmma_kernel<64, HDIM, 1, false, false><<<grows, 256, 0, stream>>>(
      inputs, wt_state, state_b, nullptr, state,
      nullptr, nullptr, nullptr, nullptr, 0);

  int kk = 0;
  for (int d = 0; d < 3; ++d) {
    const Block& B = blk[d];
    for (int it = 0; it < 5; ++it) {
      feat_build_kernel<<<NRES, HDIM, 0, stream>>>(angles, state, unl_w, unl_b, feat);

      for (int j = 0; j < 4; ++j) {
        const Inner& I = B.inner[j];
        const int dil = 1 << j;  // 1, 2, 4, 8
        // conv1: 1x1, 256 -> 64
        gn_stats_kernel<8><<<BCH * 32, 256, 0, stream>>>(feat, mu, rs);
        gemm_wmma_kernel<CDIM, HCDIM, 1, true, false><<<grows, 128, 0, stream>>>(
            feat, wt1[d][j], I.b[0], nullptr, h1, mu, rs, I.gn_g[0], I.gn_b[0], 0);
        // conv2: k=3 dilated, 64 -> 64
        gn_stats_kernel<2><<<BCH * 32, 256, 0, stream>>>(h1, mu, rs);
        gemm_wmma_kernel<HCDIM, HCDIM, 3, true, false><<<grows, 128, 0, stream>>>(
            h1, wt2[d][j], I.b[1], nullptr, h2, mu, rs, I.gn_g[1], I.gn_b[1], dil);
        // conv3: 1x1, 64 -> 256, + residual, write back to feat
        gn_stats_kernel<2><<<BCH * 32, 256, 0, stream>>>(h2, mu, rs);
        gemm_wmma_kernel<HCDIM, CDIM, 1, true, true><<<grows, 512, 0, stream>>>(
            h2, wt3[d][j], I.b[2], feat, feat, mu, rs, I.gn_g[2], I.gn_b[2], 0);
      }

      // out = feat @ aw + ab ; state = feat @ hw + hb
      gemm_wmma_kernel<CDIM, HDIM, 1, false, false><<<grows, 256, 0, stream>>>(
          feat, wt_aw[d], B.ab, nullptr, outb, nullptr, nullptr, nullptr, nullptr, 0);
      gemm_wmma_kernel<CDIM, HDIM, 1, false, false><<<grows, 256, 0, stream>>>(
          feat, wt_hw[d], B.hb, nullptr, state, nullptr, nullptr, nullptr, nullptr, 0);

      angles_update_kernel<<<NRES, HDIM, 0, stream>>>(
          outb, look_w, look_b, noise + (size_t)kk * NRES * 3, angles, aclean);
      ++kk;
    }
  }

  // distances from pre-noise angles of the last iteration
  nerf_chain_kernel<<<1, 64, 0, stream>>>(aclean, (float*)d_out);
  // returned angles are post-noise
  hipMemcpyAsync((float*)d_out + (size_t)NRES * 9, angles,
                 (size_t)NRES * 3 * sizeof(float), hipMemcpyDeviceToDevice, stream);

  (void)in_sizes; (void)n_in; (void)out_size; (void)ws_size;
}
